// Network_80942953660437
// MI455X (gfx1250) — matrix operand order repaired and checked by reading, not compiled, not run on hardware
//
#include <hip/hip_runtime.h>
#include <hip/hip_bf16.h>

#define NTOK  4096
#define CIN_  256
#define CKEY_ 448
#define NBATCH 4
#define NEGV  (-1e15f)
#define EPSV  1e-5f

typedef __attribute__((ext_vector_type(16))) __bf16 v16bf;
typedef __attribute__((ext_vector_type(8)))  __bf16 v8bf;
typedef __attribute__((ext_vector_type(8)))  float  v8f;
typedef __attribute__((ext_vector_type(4)))  float  v4f;
typedef __attribute__((ext_vector_type(4)))  unsigned int v4u;
typedef __attribute__((ext_vector_type(8)))  int v8i;
typedef __attribute__((ext_vector_type(4)))  int v4i;

#if __has_builtin(__builtin_amdgcn_tensor_load_to_lds)
#define HAVE_TDM 1
#else
#define HAVE_TDM 0
#endif

union Frag16 { v16bf v; uint4 q[2]; };
union Store8 { v8bf v; uint4 q; };

__device__ inline v16bf pack16f(const float* f) {
  v16bf r;
#pragma unroll
  for (int i = 0; i < 16; ++i) r[i] = (__bf16)f[i];
  return r;
}

__device__ inline v8f zero8() {
  v8f r;
#pragma unroll
  for (int i = 0; i < 8; ++i) r[i] = 0.0f;
  return r;
}

// ---------------------------------------------------------------------------
// Projection GEMM: Out[o, t] = sum_c W[o,c] * X[c,t] + bias[o]
// X: f32 channel-major [Cin, NTOK] per batch. Output bf16.
// tokenMajor=1 -> out[t*Cout + o] (optionally outSq = square, for Hv^2)
// tokenMajor=0 -> out[o*NTOK + t] (channel-major, for G)
// Wave tile: 16 o x 16 t. Block = 4 waves -> 16 o x 64 t.
// ---------------------------------------------------------------------------
__global__ __launch_bounds__(128)
void proj_kernel(const float* __restrict__ X, const float* __restrict__ Wm,
                 const float* __restrict__ bias,
                 __hip_bfloat16* __restrict__ outA,
                 __hip_bfloat16* __restrict__ outSq,
                 int Cin, int Cout, int tokenMajor)
{
  const int bz   = blockIdx.z;
  const int o0   = blockIdx.y * 16;
  const int lane = threadIdx.x & 31;
  const int wv   = threadIdx.x >> 5;
  const int m0   = blockIdx.x * 64 + wv * 16;
  const int hl   = (lane >> 4) & 1;   // lane half (WMMA fragment split)
  const int l15  = lane & 15;

  const float* Xb = X + (size_t)bz * Cin * NTOK;
  v8f acc = zero8();
  const int KT = Cin >> 5;
  for (int j = 0; j < KT; ++j) {
    // A fragment: row o = o0+l15; K order {0..7,16..23} (lane<16) / {8..15,24..31}
    float af[16];
    {
      const float* wrow = Wm + (size_t)(o0 + l15) * Cin + j * 32 + hl * 8;
      v4f a0 = *(const v4f*)(wrow);
      v4f a1 = *(const v4f*)(wrow + 4);
      v4f a2 = *(const v4f*)(wrow + 16);
      v4f a3 = *(const v4f*)(wrow + 20);
#pragma unroll
      for (int i = 0; i < 4; ++i) { af[i]=a0[i]; af[4+i]=a1[i]; af[8+i]=a2[i]; af[12+i]=a3[i]; }
    }
    // B fragment: column = token m0+l15, K order same as A
    float bfv[16];
    {
      const float* xcol = Xb + (size_t)(j * 32 + hl * 8) * NTOK + m0 + l15;
#pragma unroll
      for (int i = 0; i < 16; ++i) bfv[i] = xcol[(size_t)(i < 8 ? i : i + 8) * NTOK];
    }
    v16bf Av = pack16f(af);
    v16bf Bv = pack16f(bfv);
    acc = __builtin_amdgcn_wmma_f32_16x16x32_bf16(false, Av, false, Bv, (short)0, acc,
                                                  false, false);
  }

  const int m = m0 + l15;
  float res[8];
#pragma unroll
  for (int r = 0; r < 8; ++r)
    res[r] = acc[r] + bias[o0 + hl * 8 + r];

  if (tokenMajor) {
    Store8 st;
#pragma unroll
    for (int r = 0; r < 8; ++r) st.v[r] = (__bf16)res[r];
    *(uint4*)(outA + ((size_t)bz * NTOK + m) * Cout + o0 + hl * 8) = st.q;
    if (outSq) {
      Store8 s2;
#pragma unroll
      for (int r = 0; r < 8; ++r) s2.v[r] = (__bf16)(res[r] * res[r]);
      *(uint4*)(outSq + ((size_t)bz * NTOK + m) * Cout + o0 + hl * 8) = s2.q;
    }
  } else {
    __hip_bfloat16* ob = outA + (size_t)bz * Cout * NTOK;
#pragma unroll
    for (int r = 0; r < 8; ++r)
      ob[(size_t)(o0 + hl * 8 + r) * NTOK + m] = __float2bfloat16(res[r]);
  }
}

// ---------------------------------------------------------------------------
// Fused flash-attention: per (batch, 16-row block), online softmax over 4096
// style tokens in tiles of 32; accumulates A@Hv and A@Hv2 in f32 fragments.
// Block = 4 waves; each wave owns 64 channels (4 c-tiles x {mean,m2}).
// Fq row block staged into LDS by the Tensor Data Mover (TDM) when available.
// ---------------------------------------------------------------------------
__global__ __launch_bounds__(128)
void attn_kernel(const __hip_bfloat16* __restrict__ Fq,  // [B][N][CKEY] token-major
                 const __hip_bfloat16* __restrict__ G,   // [B][CKEY][N] channel-major
                 const __hip_bfloat16* __restrict__ Hv,  // [B][N][CIN] token-major
                 const __hip_bfloat16* __restrict__ Hv2, // [B][N][CIN]
                 const int* __restrict__ cmask, const int* __restrict__ smask,
                 float* __restrict__ meanW, float* __restrict__ stdW) // [B][N][CIN]
{
  const int b    = blockIdx.x >> 8;        // N/16 = 256 row-blocks per batch
  const int n0   = (blockIdx.x & 255) * 16;
  const int lane = threadIdx.x & 31;
  const int wv   = threadIdx.x >> 5;
  const int hl   = (lane >> 4) & 1;
  const int l15  = lane & 15;
  const int c0   = wv * 64;

  __shared__ __hip_bfloat16 FqLds[16 * CKEY_];   // 14336 B, staged once
  __shared__ float Pbuf[4][16][32];              // per-wave P transpose bounce

#if HAVE_TDM
  if (threadIdx.x < 32) {
    // TDM descriptor: 2D tile, 16 rows x 448 bf16, row stride 448 elements.
    const unsigned long long ga =
        (unsigned long long)(const void*)(Fq + ((size_t)b * NTOK + n0) * CKEY_);
    const unsigned ldsa = (unsigned)(unsigned long long)(void*)FqLds;
    v4u g0 = { 1u,                                  // count=1, user descriptor
               ldsa,                                 // lds_addr [63:32]
               (unsigned)ga,                         // global_addr[31:0]
               (unsigned)(ga >> 32) | (2u << 30) };  // global_addr[56:32] | type=2
    v8i g1 = { (int)(1u << 16),            // workgroup_mask=0, data_size=2B (enc 1)
               (int)(448u << 16),          // tensor_dim0[15:0] << 16
               (int)(16u << 16),           // tensor_dim0[31:16]=0 | tensor_dim1[15:0]
               (int)(448u << 16),          // tensor_dim1[31:16]=0 | tile_dim0=448
               16,                         // tile_dim1=16, tile_dim2=0
               448,                        // tensor_dim0_stride[31:0]
               0, 0 };                     // stride hi / dim1 stride (unused, 2D)
    v4i z4 = { 0, 0, 0, 0 };
#if defined(__clang_major__) && (__clang_major__ >= 23)
    v8i z8 = { 0, 0, 0, 0, 0, 0, 0, 0 };
    __builtin_amdgcn_tensor_load_to_lds(g0, g1, z4, z4, z8, 0);
#else
    __builtin_amdgcn_tensor_load_to_lds(g0, g1, z4, z4, 0);
#endif
#if __has_builtin(__builtin_amdgcn_s_wait_tensorcnt)
    __builtin_amdgcn_s_wait_tensorcnt(0);
#endif
  }
  __syncthreads();
#else
  { // cooperative stage of the 16 query rows (contiguous token-major)
    const uint4* src = (const uint4*)(Fq + ((size_t)b * NTOK + n0) * CKEY_);
    uint4* dst = (uint4*)FqLds;
    for (int i = threadIdx.x; i < (16 * CKEY_ * 2) / 16; i += 128) dst[i] = src[i];
  }
  __syncthreads();
#endif

  int cmv[8];
#pragma unroll
  for (int r = 0; r < 8; ++r)
    cmv[r] = cmask[(size_t)b * NTOK + n0 + hl * 8 + r];

  float mr[8], Lr[8];
#pragma unroll
  for (int r = 0; r < 8; ++r) { mr[r] = -3e38f; Lr[r] = 0.0f; }
  v8f accM[4], accV[4];
#pragma unroll
  for (int ct = 0; ct < 4; ++ct) { accM[ct] = zero8(); accV[ct] = zero8(); }

  const __hip_bfloat16* Gb  = G   + (size_t)b * CKEY_ * NTOK;
  const __hip_bfloat16* Hb  = Hv  + (size_t)b * NTOK * CIN_;
  const __hip_bfloat16* H2b = Hv2 + (size_t)b * NTOK * CIN_;
  const int* smb = smask + (size_t)b * NTOK;
  float* myP = &Pbuf[wv][0][0];

  for (int m0 = 0; m0 < NTOK; m0 += 32) {
    // ---- S tile (16 rows x 32 style cols), K = 448 over 14 WMMA steps ----
    v8f s0 = zero8(), s1 = zero8();
#pragma unroll
    for (int j = 0; j < 14; ++j) {
      Frag16 A;
      const char* ab = (const char*)FqLds + l15 * (CKEY_ * 2) + (j * 32 + hl * 8) * 2;
      A.q[0] = *(const uint4*)(ab);
      A.q[1] = *(const uint4*)(ab + 32);
      const __bf16* gb = (const __bf16*)(Gb + (size_t)(j * 32 + hl * 8) * NTOK + m0 + l15);
      __builtin_prefetch(gb + 32, 0, 2);  // next m-tile, same channel row
      Frag16 B0, B1;
#pragma unroll
      for (int i = 0; i < 16; ++i) {
        const size_t ko = (size_t)(i < 8 ? i : i + 8) * NTOK;
        B0.v[i] = gb[ko]; B1.v[i] = gb[ko + 16];
      }
      s0 = __builtin_amdgcn_wmma_f32_16x16x32_bf16(false, A.v, false, B0.v, (short)0, s0,
                                                   false, false);
      s1 = __builtin_amdgcn_wmma_f32_16x16x32_bf16(false, A.v, false, B1.v, (short)0, s1,
                                                   false, false);
    }

    // ---- mask + online softmax (row stats live per 16-lane half) ----
    const int sm0 = smb[m0 + l15];
    const int sm1 = smb[m0 + 16 + l15];
    float p0[8], p1[8];
#pragma unroll
    for (int r = 0; r < 8; ++r) {
      float v0 = (cmv[r] && sm0 == 0) ? NEGV : s0[r];
      float v1 = (cmv[r] && sm1 == 0) ? NEGV : s1[r];
      float t = fmaxf(v0, v1);
      t = fmaxf(t, __shfl_xor(t, 1, 32));
      t = fmaxf(t, __shfl_xor(t, 2, 32));
      t = fmaxf(t, __shfl_xor(t, 4, 32));
      t = fmaxf(t, __shfl_xor(t, 8, 32));
      float mn = fmaxf(mr[r], t);
      float al = __expf(mr[r] - mn);
      mr[r] = mn;
      p0[r] = __expf(v0 - mn);
      p1[r] = __expf(v1 - mn);
      float u = p0[r] + p1[r];
      u += __shfl_xor(u, 1, 32);
      u += __shfl_xor(u, 2, 32);
      u += __shfl_xor(u, 4, 32);
      u += __shfl_xor(u, 8, 32);
      Lr[r] = Lr[r] * al + u;
#pragma unroll
      for (int ct = 0; ct < 4; ++ct) { accM[ct][r] *= al; accV[ct][r] *= al; }
    }

    // ---- transpose P (D-layout) -> A-fragment via wave-private LDS ----
#pragma unroll
    for (int r = 0; r < 8; ++r) {
      myP[(hl * 8 + r) * 32 + l15]      = p0[r];
      myP[(hl * 8 + r) * 32 + l15 + 16] = p1[r];
    }
    Frag16 Ap;
    {
      const float* pr = myP + l15 * 32 + hl * 8;
      v4f f0 = *(const v4f*)(pr);
      v4f f1 = *(const v4f*)(pr + 4);
      v4f f2 = *(const v4f*)(pr + 16);
      v4f f3 = *(const v4f*)(pr + 20);
      float tf[16];
#pragma unroll
      for (int i = 0; i < 4; ++i) { tf[i]=f0[i]; tf[4+i]=f1[i]; tf[8+i]=f2[i]; tf[12+i]=f3[i]; }
      Ap.v = pack16f(tf);
    }

    // ---- accumulate mean and m2 fragments for this wave's 64 channels ----
#pragma unroll
    for (int ct = 0; ct < 4; ++ct) {
      const __bf16* hb = (const __bf16*)(Hb + (size_t)(m0 + hl * 8) * CIN_ + c0 + ct * 16 + l15);
      Frag16 Bm;
#pragma unroll
      for (int i = 0; i < 16; ++i) Bm.v[i] = hb[(size_t)(i < 8 ? i : i + 8) * CIN_];
      accM[ct] = __builtin_amdgcn_wmma_f32_16x16x32_bf16(false, Ap.v, false, Bm.v, (short)0,
                                                         accM[ct], false, false);
      const __bf16* h2 = (const __bf16*)(H2b + (size_t)(m0 + hl * 8) * CIN_ + c0 + ct * 16 + l15);
      Frag16 B2;
#pragma unroll
      for (int i = 0; i < 16; ++i) B2.v[i] = h2[(size_t)(i < 8 ? i : i + 8) * CIN_];
      accV[ct] = __builtin_amdgcn_wmma_f32_16x16x32_bf16(false, Ap.v, false, B2.v, (short)0,
                                                         accV[ct], false, false);
    }
  }

  // ---- finalize: mean = acc/L, std = sqrt(relu(m2/L - mean^2)) ----
#pragma unroll
  for (int r = 0; r < 8; ++r) {
    float rl = 1.0f / Lr[r];
    const int n = n0 + hl * 8 + r;
#pragma unroll
    for (int ct = 0; ct < 4; ++ct) {
      float mn = accM[ct][r] * rl;
      float m2 = accV[ct][r] * rl;
      float sd = sqrtf(fmaxf(m2 - mn * mn, 0.0f));
      size_t idx = ((size_t)b * NTOK + n) * CIN_ + c0 + ct * 16 + l15;
      meanW[idx] = mn;
      stdW[idx]  = sd;
    }
  }
}

// ---------------------------------------------------------------------------
// Per-(batch,channel) mean / inv-std of content (unbiased var), 1 block each.
// ---------------------------------------------------------------------------
__global__ __launch_bounds__(256)
void mvn_stats_kernel(const float* __restrict__ content, float* __restrict__ stats)
{
  const int bc = blockIdx.x;
  const float* x = content + (size_t)bc * NTOK;
  float s = 0.0f, s2 = 0.0f;
  for (int i = threadIdx.x; i < NTOK; i += 256) { float v = x[i]; s += v; s2 += v * v; }
  __shared__ float sh[256], sh2[256];
  sh[threadIdx.x] = s; sh2[threadIdx.x] = s2;
  __syncthreads();
  for (int off = 128; off > 0; off >>= 1) {
    if (threadIdx.x < off) { sh[threadIdx.x] += sh[threadIdx.x + off];
                             sh2[threadIdx.x] += sh2[threadIdx.x + off]; }
    __syncthreads();
  }
  if (threadIdx.x == 0) {
    float mean = sh[0] / (float)NTOK;
    float var  = (sh2[0] - sh[0] * sh[0] / (float)NTOK) / (float)(NTOK - 1);
    stats[2 * bc]     = mean;
    stats[2 * bc + 1] = rsqrtf(var + EPSV);
  }
}

// ---------------------------------------------------------------------------
// out[b,c,n] = std[b,n,c] * ((content - mu_c) * rstd_c) + mean[b,n,c]
// ---------------------------------------------------------------------------
__global__ __launch_bounds__(256)
void final_kernel(const float* __restrict__ content, const float* __restrict__ stats,
                  const float* __restrict__ meanW, const float* __restrict__ stdW,
                  float* __restrict__ out)
{
  size_t i = (size_t)blockIdx.x * 256 + threadIdx.x;
  const size_t total = (size_t)NBATCH * CIN_ * NTOK;
  if (i >= total) return;
  const int n  = (int)(i % NTOK);
  const size_t bc = i / NTOK;
  const float mu = stats[2 * bc];
  const float rs = stats[2 * bc + 1];
  const int c = (int)(bc % CIN_);
  const int b = (int)(bc / CIN_);
  const size_t t = ((size_t)b * NTOK + n) * CIN_ + c;
  out[i] = stdW[t] * ((content[i] - mu) * rs) + meanW[t];
}

// ---------------------------------------------------------------------------
extern "C" void kernel_launch(void* const* d_in, const int* in_sizes, int n_in,
                              void* d_out, int out_size, void* d_ws, size_t ws_size,
                              hipStream_t stream)
{
  const float* content = (const float*)d_in[0];
  const float* style   = (const float*)d_in[1];
  const float* ckey    = (const float*)d_in[2];
  const float* skey    = (const float*)d_in[3];
  const int*   cmask   = (const int*)d_in[4];
  const int*   smask   = (const int*)d_in[5];
  const float* Wf = (const float*)d_in[6];
  const float* bf = (const float*)d_in[7];
  const float* Wg = (const float*)d_in[8];
  const float* bg = (const float*)d_in[9];
  const float* Wh = (const float*)d_in[10];
  const float* bh = (const float*)d_in[11];
  float* out = (float*)d_out;

  char* ws = (char*)d_ws;
  size_t off = 0;
  auto take = [&](size_t bytes) { void* p = ws + off; off += (bytes + 255) & ~(size_t)255; return p; };
  __hip_bfloat16* FqW  = (__hip_bfloat16*)take((size_t)NBATCH * NTOK * CKEY_ * 2);
  __hip_bfloat16* GW   = (__hip_bfloat16*)take((size_t)NBATCH * CKEY_ * NTOK * 2);
  __hip_bfloat16* HvW  = (__hip_bfloat16*)take((size_t)NBATCH * NTOK * CIN_ * 2);
  __hip_bfloat16* Hv2W = (__hip_bfloat16*)take((size_t)NBATCH * NTOK * CIN_ * 2);
  float* meanW = (float*)take((size_t)NBATCH * NTOK * CIN_ * 4);
  float* stdW  = (float*)take((size_t)NBATCH * NTOK * CIN_ * 4);
  float* stats = (float*)take((size_t)NBATCH * CIN_ * 2 * 4);
  (void)ws_size; (void)in_sizes; (void)n_in; (void)out_size;

  // Projections (WMMA GEMMs)
  proj_kernel<<<dim3(NTOK / 64, CKEY_ / 16, NBATCH), 128, 0, stream>>>(
      ckey, Wf, bf, FqW, nullptr, CKEY_, CKEY_, 1);          // Fq token-major
  proj_kernel<<<dim3(NTOK / 64, CKEY_ / 16, NBATCH), 128, 0, stream>>>(
      skey, Wg, bg, GW, nullptr, CKEY_, CKEY_, 0);           // G channel-major
  proj_kernel<<<dim3(NTOK / 64, CIN_ / 16, NBATCH), 128, 0, stream>>>(
      style, Wh, bh, HvW, Hv2W, CIN_, CIN_, 1);              // Hv + Hv^2 token-major

  // Fused flash attention -> mean/std per token
  attn_kernel<<<NBATCH * (NTOK / 16), 128, 0, stream>>>(
      FqW, GW, HvW, Hv2W, cmask, smask, meanW, stdW);

  // mvn stats + final combine
  mvn_stats_kernel<<<NBATCH * CIN_, 256, 0, stream>>>(content, stats);
  final_kernel<<<(int)(((size_t)NBATCH * CIN_ * NTOK + 255) / 256), 256, 0, stream>>>(
      content, stats, meanW, stdW, out);
}
